// WaveCRN_33638183862756
// MI455X (gfx1250) — matrix operand order repaired and checked by reading, not compiled, not run on hardware
//
#include <hip/hip_runtime.h>
#include <hip/hip_bf16.h>

// ---------------------------------------------------------------------------
// WaveCRN for MI455X (gfx1250, wave32, WMMA).
// GEMMs (150 GFLOP) on v_wmma_f32_16x16x32_bf16, f32 accumulate.
// Weights pre-swizzled once per launch into per-lane fragment order so B
// fragments come straight from global/L2; the activation tile (shared by all
// 8 waves) is double-buffered in LDS: one barrier per 32-deep K chunk.
// ---------------------------------------------------------------------------

#define L_SEQ   1025
#define BATCH   16
#define MROWS   (L_SEQ * BATCH)       // 16400
#define MPAD    16448                 // 257 * 64  (GEMM block tile M=64)
#define OUTW    512
#define TLEN    49152

typedef float  v8f   __attribute__((ext_vector_type(8)));
typedef __bf16 v16bf __attribute__((ext_vector_type(16)));
typedef __bf16 v8bf  __attribute__((ext_vector_type(8)));

// float -> bf16 bits, round-to-nearest-even
__device__ __forceinline__ unsigned short f2bf(float x) {
    union { float f; unsigned u; } c; c.f = x;
    unsigned u = c.u;
    if ((u & 0x7fffffffu) > 0x7f800000u)          // NaN
        return (unsigned short)((u >> 16) | 0x40);
    unsigned r = u + 0x7fffu + ((u >> 16) & 1u);
    return (unsigned short)(r >> 16);
}

__device__ __forceinline__ float sigmoidf(float x) {
    return 1.0f / (1.0f + __expf(-x));
}

// ---------------------------------------------------------------------------
// Elementwise helpers
// ---------------------------------------------------------------------------
__global__ __launch_bounds__(256)
void k_f32_to_bf16(const float* __restrict__ src, unsigned short* __restrict__ dst, long n) {
    long i = (long)blockIdx.x * 256 + threadIdx.x;
    if (i < n) dst[i] = f2bf(src[i]);
}

__global__ __launch_bounds__(256)
void k_zero_bf16(unsigned short* __restrict__ dst, long n) {
    long i = (long)blockIdx.x * 256 + threadIdx.x;
    if (i < n) dst[i] = 0;
}

__global__ __launch_bounds__(256)
void k_gate(float* __restrict__ o, const float* __restrict__ hin, long n) {
    long i = (long)blockIdx.x * 256 + threadIdx.x;
    if (i < n) o[i] *= hin[i];
}

// ---------------------------------------------------------------------------
// Weight pre-swizzle: source B[k][n] (or transposed source when trans=1,
// i.e. B[k][n] = src[n*K + k]) -> packed per-lane WMMA fragment order:
//   P[ (nt*(K/32)+kt)*512 + lane*16 + j ]  with
//   n = nt*16 + (lane&15),  k = kt*32 + (lane>>4)*16 + j
// A wave then loads its 16x32 B fragment as two coalesced 16B loads per lane.
// ---------------------------------------------------------------------------
__global__ __launch_bounds__(256)
void k_pack_b(const float* __restrict__ src, unsigned short* __restrict__ P,
              int K, int N, int trans) {
    long idx = (long)blockIdx.x * 256 + threadIdx.x;
    if (idx >= (long)K * N) return;
    int j  = (int)(idx & 15);
    int ln = (int)((idx >> 4) & 31);
    long t = idx >> 9;                 // nt*(K/32) + kt
    int kt = (int)(t % (K / 32));
    int nt = (int)(t / (K / 32));
    int n = nt * 16 + (ln & 15);
    int k = kt * 32 + (ln >> 4) * 16 + j;
    float v = trans ? src[(size_t)n * K + k] : src[(size_t)k * N + n];
    P[idx] = f2bf(v);
}

// ---------------------------------------------------------------------------
// Conv1d front end: x[16,1,49152] * w[256,1,96], stride 48, pad 48
// ---------------------------------------------------------------------------
__global__ __launch_bounds__(256)
void k_conv_front(const float* __restrict__ x, const float* __restrict__ w,
                  float* __restrict__ h) {
    __shared__ float xs[96];
    const int lb = blockIdx.x;              // l*16 + b
    const int l = lb >> 4, b = lb & 15;
    const int t0 = l * 48 - 48;
    if (threadIdx.x < 96) {
        int p = t0 + (int)threadIdx.x;
        xs[threadIdx.x] = (p >= 0 && p < TLEN) ? x[(size_t)b * TLEN + p] : 0.0f;
    }
    __syncthreads();
    const int c = threadIdx.x;
    const float* wr = w + c * 96;
    float acc = 0.0f;
#pragma unroll 8
    for (int j = 0; j < 96; ++j) acc += xs[j] * wr[j];
    h[(size_t)lb * 256 + c] = acc;
}

// ---------------------------------------------------------------------------
// LayerNorm (f32 in) -> bf16 out; one block per row
// ---------------------------------------------------------------------------
__global__ __launch_bounds__(256)
void k_ln_bf16(const float* __restrict__ y, int n, const float* __restrict__ g,
               const float* __restrict__ be, unsigned short* __restrict__ A) {
    __shared__ float s1[256], s2[256];
    const int row = blockIdx.x;
    const float* xr = y + (size_t)row * n;
    float a = 0.0f, b = 0.0f;
    for (int i = threadIdx.x; i < n; i += 256) { float v = xr[i]; a += v; b += v * v; }
    s1[threadIdx.x] = a; s2[threadIdx.x] = b;
    __syncthreads();
    for (int off = 128; off > 0; off >>= 1) {
        if ((int)threadIdx.x < off) {
            s1[threadIdx.x] += s1[threadIdx.x + off];
            s2[threadIdx.x] += s2[threadIdx.x + off];
        }
        __syncthreads();
    }
    const float mean = s1[0] / (float)n;
    const float var  = s2[0] / (float)n - mean * mean;
    const float rstd = rsqrtf(var + 1e-5f);
    for (int i = threadIdx.x; i < n; i += 256)
        A[(size_t)row * n + i] = f2bf((xr[i] - mean) * rstd * g[i] + be[i]);
}

// ---------------------------------------------------------------------------
// bf16 WMMA GEMM: C[MPAD,N] = A[MPAD,K](bf16) * Bpack(bf16), f32 accumulate.
// Block tile 64(M) x 256(N), 8 waves; each wave owns 64x32 (4 M-tiles x
// 2 N-tiles = 8 accumulators, 8 WMMAs per 32-deep K chunk).
// A is double-buffered in LDS (one barrier per chunk, staging of chunk kt+1
// overlaps the 8 WMMAs of chunk kt); B fragments are loaded directly from
// the pre-swizzled global weight buffer (L2 resident, coalesced 16B/lane).
// ---------------------------------------------------------------------------
template <int K, int N>
__global__ __launch_bounds__(256)
void k_gemm_bf16_wmma(const unsigned short* __restrict__ A,
                      const unsigned short* __restrict__ Bp,
                      float* __restrict__ C) {
    __shared__ unsigned short As[2][64 * 32];   // [buf][m][k]
    constexpr int KT = K / 32;
    const int tid   = threadIdx.x;
    const int lane  = tid & 31;
    const int wv    = tid >> 5;               // 0..7
    const int m0    = blockIdx.y * 64;
    const int nblk  = blockIdx.x;             // N/256 blocks
    const int mh    = lane & 15;
    const int khalf = lane >> 4;
    const int nt0   = nblk * 16 + wv * 2;     // two 16-col tiles per wave

    // A staging addresses (fixed per thread)
    const int arow = tid >> 2;
    const int ach  = (tid & 3) << 3;
    const unsigned short* agp = A + (size_t)(m0 + arow) * K + ach;

    v8f acc[4][2];
#pragma unroll
    for (int i = 0; i < 4; ++i)
#pragma unroll
        for (int t = 0; t < 2; ++t)
#pragma unroll
            for (int j = 0; j < 8; ++j) acc[i][t][j] = 0.0f;

    // prologue: stage chunk 0
    *reinterpret_cast<uint4*>(&As[0][arow * 32 + ach]) =
        *reinterpret_cast<const uint4*>(agp);
    __builtin_prefetch(agp + 32, 0, 3);       // near prefetch of chunk 1
    __syncthreads();

    for (int kt = 0; kt < KT; ++kt) {
        const int cur = kt & 1;
        // --- stage next A chunk into the other buffer (overlaps WMMAs) ---
        if (kt + 1 < KT) {
            *reinterpret_cast<uint4*>(&As[cur ^ 1][arow * 32 + ach]) =
                *reinterpret_cast<const uint4*>(agp + (size_t)(kt + 1) * 32);
            if (kt + 2 < KT)
                __builtin_prefetch(agp + (size_t)(kt + 2) * 32, 0, 3);
        }

        // --- B fragments straight from packed global (2 x 32B per lane) ---
        union { v16bf v; v8bf h[2]; } bf0, bf1;
        {
            const unsigned short* p0 = Bp + ((size_t)(nt0 * KT + kt) * 512) + lane * 16;
            const unsigned short* p1 = p0 + (size_t)KT * 512;   // next n-tile
            bf0.h[0] = *reinterpret_cast<const v8bf*>(p0);
            bf0.h[1] = *reinterpret_cast<const v8bf*>(p0 + 8);
            bf1.h[0] = *reinterpret_cast<const v8bf*>(p1);
            bf1.h[1] = *reinterpret_cast<const v8bf*>(p1 + 8);
        }
        // --- 4 A fragments, each reused for 2 WMMAs ---
        const unsigned short* asb = &As[cur][0];
#pragma unroll
        for (int mt = 0; mt < 4; ++mt) {
            const int mr = mt * 16 + mh;
            union { v16bf v; v8bf h[2]; } af;   // K = {khalf*16..+15}
            af.h[0] = *reinterpret_cast<const v8bf*>(asb + mr * 32 + khalf * 16);
            af.h[1] = *reinterpret_cast<const v8bf*>(asb + mr * 32 + khalf * 16 + 8);
            acc[mt][0] = __builtin_amdgcn_wmma_f32_16x16x32_bf16(
                false, af.v, false, bf0.v, (short)0, acc[mt][0], false, false);
            acc[mt][1] = __builtin_amdgcn_wmma_f32_16x16x32_bf16(
                false, af.v, false, bf1.v, (short)0, acc[mt][1], false, false);
        }
        __syncthreads();   // staging of kt+1 complete + reads of As[cur] done
    }

    // --- store: D lane = col (lane&15), rows r + 8*(lane>>4) ---
    const int nc    = nblk * 256 + wv * 32 + mh;
    const int rbase = m0 + (khalf << 3);
#pragma unroll
    for (int mt = 0; mt < 4; ++mt) {
        float* crow = C + (size_t)(rbase + mt * 16) * N + nc;
#pragma unroll
        for (int r = 0; r < 8; ++r) {
            crow[(size_t)r * N]      = acc[mt][0][r];
            crow[(size_t)r * N + 16] = acc[mt][1][r];
        }
    }
}

// ---------------------------------------------------------------------------
// Bidirectional SRU scan. U[row][ (dir*256+d)*k + j ] with row = l*16+b.
// 8192 independent channels (b,dir,d); backward dir walks time in reverse
// (equivalent to flip/scan/flip of the reference).
// ---------------------------------------------------------------------------
__global__ __launch_bounds__(256)
void k_sru_scan(const float* __restrict__ U, int N, int k,
                const float* __restrict__ v, const float* __restrict__ bias,
                const float* __restrict__ res_src,   // null when k==4
                float* __restrict__ hout) {
    const int tid = blockIdx.x * 256 + threadIdx.x;   // < 8192
    const int b   = tid >> 9;
    const int dir = (tid >> 8) & 1;
    const int d   = tid & 255;
    const int ch  = (dir << 8) + d;                   // 0..511
    const float vf = v[ch],        vr = v[512 + ch];
    const float bf = bias[ch],     br = bias[512 + ch];
    float c = 0.0f;
    for (int step = 0; step < L_SEQ; ++step) {
        const int l   = dir ? (L_SEQ - 1 - step) : step;
        const int row = l * BATCH + b;
        const float* up = U + (size_t)row * N + ch * k;
        const float xt = up[0], fp = up[1], rp = up[2];
        const float res = (k == 4) ? up[3] : res_src[(size_t)row * OUTW + ch];
        const float f = sigmoidf(fp + vf * c + bf);
        c = f * c + (1.0f - f) * xt;
        const float r = sigmoidf(rp + vr * c + br);
        hout[(size_t)row * OUTW + ch] = r * c + (1.0f - r) * res;
    }
}

// ---------------------------------------------------------------------------
// ConvTranspose1d(256->1, k=96, s=48, p=48) + tanh.
// ---------------------------------------------------------------------------
__global__ __launch_bounds__(256)
void k_deconv_tanh(const float* __restrict__ O, const float* __restrict__ dw,
                   float* __restrict__ out) {
    const int idx = blockIdx.x * 256 + threadIdx.x;   // < 16*49152
    const int b = idx / TLEN;
    const int t = idx - b * TLEN;
    const int lh = (t + 48) / 48;                     // 1..1024
    const int kh = t + 48 - lh * 48;                  // 0..47
    const float* o1 = O + (size_t)(lh * BATCH + b) * 256;
    const float* o0 = O + (size_t)((lh - 1) * BATCH + b) * 256;
    float acc = 0.0f;
#pragma unroll 4
    for (int c = 0; c < 256; ++c)
        acc += o1[c] * dw[c * 96 + kh] + o0[c] * dw[c * 96 + kh + 48];
    out[idx] = tanhf(acc);
}

// ---------------------------------------------------------------------------
// Launch
// ---------------------------------------------------------------------------
extern "C" void kernel_launch(void* const* d_in, const int* in_sizes, int n_in,
                              void* d_out, int out_size, void* d_ws, size_t ws_size,
                              hipStream_t stream) {
    const float* x      = (const float*)d_in[0];
    const float* conv_w = (const float*)d_in[1];
    const float* dec_w  = (const float*)d_in[2];
    const float* fc_w   = (const float*)d_in[3];
    const float* W0     = (const float*)d_in[4];
    const float* v0     = (const float*)d_in[5];
    const float* b0     = (const float*)d_in[6];
    const float* g0     = (const float*)d_in[7];
    const float* be0    = (const float*)d_in[8];
    const float* Ws     = (const float*)d_in[9];
    const float* vs     = (const float*)d_in[10];
    const float* bs     = (const float*)d_in[11];
    const float* gs     = (const float*)d_in[12];
    const float* bes    = (const float*)d_in[13];

    // ---- workspace carving (≈245 MB) ----
    char* ws = (char*)d_ws;
    size_t off = 0;
    auto carve = [&](size_t bytes) { char* p = ws + off; off = (off + bytes + 255) & ~(size_t)255; return p; };
    unsigned short* Abf  = (unsigned short*)carve((size_t)MPAD * 512 * 2);   // LN / y in bf16
    float*          hin  = (float*)carve((size_t)MPAD * 256 * 4);            // conv output (f32)
    float*          yA   = (float*)carve((size_t)MPAD * 512 * 4);
    float*          yB   = (float*)carve((size_t)MPAD * 512 * 4);
    float*          U    = (float*)carve((size_t)MPAD * 2048 * 4);           // GEMM out (O aliases)
    unsigned short* Wb0p = (unsigned short*)carve((size_t)256 * 2048 * 2);   // packed fragments
    unsigned short* Wsbp = (unsigned short*)carve((size_t)5 * 512 * 1536 * 2);
    unsigned short* fcTp = (unsigned short*)carve((size_t)512 * 256 * 2);
    float*          Obuf = U;                                                // FC output alias

    // ---- weight pre-swizzle into WMMA fragment order (deterministic, per call) ----
    k_pack_b<<<(256 * 2048) / 256, 256, 0, stream>>>(W0, Wb0p, 256, 2048, 0);
    for (int i = 0; i < 5; ++i)
        k_pack_b<<<(512 * 1536) / 256, 256, 0, stream>>>(
            Ws + (size_t)i * 512 * 1536, Wsbp + (size_t)i * 512 * 1536, 512, 1536, 0);
    k_pack_b<<<(512 * 256) / 256, 256, 0, stream>>>(fc_w, fcTp, 512, 256, 1);
    // zero bf16 pad rows for both row strides (256 for layer0, 512 afterwards)
    k_zero_bf16<<<((MPAD - MROWS) * 256) / 256, 256, 0, stream>>>(Abf + (size_t)MROWS * 256, (long)(MPAD - MROWS) * 256);
    k_zero_bf16<<<((MPAD - MROWS) * 512) / 256, 256, 0, stream>>>(Abf + (size_t)MROWS * 512, (long)(MPAD - MROWS) * 512);

    // ---- conv front end ----
    k_conv_front<<<MROWS, 256, 0, stream>>>(x, conv_w, hin);

    // ---- SRU layer 0 (n_in=256, k=4, N=2048) ----
    k_ln_bf16<<<MROWS, 256, 0, stream>>>(hin, 256, g0, be0, Abf);
    {
        dim3 grid(2048 / 256, MPAD / 64);
        k_gemm_bf16_wmma<256, 2048><<<grid, 256, 0, stream>>>(Abf, Wb0p, U);
    }
    k_sru_scan<<<8192 / 256, 256, 0, stream>>>(U, 2048, 4, v0, b0, nullptr, yA);

    // ---- SRU layers 1..5 (n_in=512, k=3, N=1536), ping-pong ----
    float* yin = yA; float* yout = yB;
    for (int i = 0; i < 5; ++i) {
        k_ln_bf16<<<MROWS, 256, 0, stream>>>(yin, 512, gs + i * 512, bes + i * 512, Abf);
        dim3 grid(1536 / 256, MPAD / 64);
        k_gemm_bf16_wmma<512, 1536><<<grid, 256, 0, stream>>>(
            Abf, Wsbp + (size_t)i * 512 * 1536, U);
        k_sru_scan<<<8192 / 256, 256, 0, stream>>>(U, 1536, 3, vs + i * 1024, bs + i * 1024, yin, yout);
        float* tmp = yin; yin = yout; yout = tmp;
    }

    // ---- FC (y @ fc_w.T) + gating ----
    k_f32_to_bf16<<<((long)MROWS * 512) / 256, 256, 0, stream>>>(yin, Abf, (long)MROWS * 512);
    {
        dim3 grid(1, MPAD / 64);
        k_gemm_bf16_wmma<512, 256><<<grid, 256, 0, stream>>>(Abf, fcTp, Obuf);
    }
    k_gate<<<((long)MROWS * 256) / 256, 256, 0, stream>>>(Obuf, hin, (long)MROWS * 256);

    // ---- deconv + tanh -> d_out ----
    k_deconv_tanh<<<(BATCH * TLEN) / 256, 256, 0, stream>>>(Obuf, dec_w, (float*)d_out);

    (void)in_sizes; (void)n_in; (void)out_size; (void)ws_size;
}
